// GCN_46978352283765
// MI455X (gfx1250) — matrix operand order repaired and checked by reading, not compiled, not run on hardware
//
#include <hip/hip_runtime.h>
#include <math.h>

// ---------------- problem constants (from reference) ----------------
#define NNODES 200000
#define NEDGES 3200000
#define FDIM   256

// ---------------- WMMA vector types (CDNA5 / gfx1250) ----------------
typedef __attribute__((ext_vector_type(16))) __bf16          v16bf;
typedef __attribute__((ext_vector_type(8)))  float           v8f;

struct BFrag32B { uint4 lo, hi; };   // 32 bytes == one v16bf fragment

// float -> bf16 bits with round-to-nearest-even (used only in one-time prepack)
static __device__ __forceinline__ unsigned int f2bf_bits(float f) {
    union { float f; unsigned int u; } c; c.f = f;
    unsigned int u = c.u;
    u += 0x7FFFu + ((u >> 16) & 1u);
    return u >> 16;
}

// ---------------- utility kernels ----------------
__global__ void __launch_bounds__(256) zero_kernel(float* __restrict__ p, long n) {
    long i = (long)blockIdx.x * 256 + threadIdx.x;
    if (i < n) p[i] = 0.0f;
}

__global__ void __launch_bounds__(256) deg_kernel(const int* __restrict__ ei,
                                                  float* __restrict__ deg) {
    int e = blockIdx.x * 256 + threadIdx.x;
    if (e >= NEDGES) return;
    atomicAdd(&deg[ei[NEDGES + e]], 1.0f);
}

__global__ void __launch_bounds__(256) rsqrt_kernel(float* __restrict__ deg) {
    int i = blockIdx.x * 256 + threadIdx.x;
    if (i < NNODES) deg[i] = rsqrtf(deg[i] + 1.0f);
}

// ---------------- layer 1 GEMM: h = x @ W1  ([N,256]x[256,32]) ----------------
// One wave per 16-row tile; output tile 16x32 = two 16x16 f32 accumulators.
// W1 is pre-packed into the exact WMMA-B bf16 fragment layout in LDS, so the
// streaming loop is: 4 global b128 (A, f32) + packed cvt + 4 ds b128 (B) + 2 WMMA.
__global__ void __launch_bounds__(256)
gemm1_wmma(const float* __restrict__ x, const float* __restrict__ W1,
           float* __restrict__ h) {
    // sB[((t*2 + c)*2 + lohi)*32 + lane] : 8 k-steps x 2 col-tiles x lo/hi x 32 lanes
    __shared__ uint4 sB[8 * 2 * 2 * 32];          // 16 KB

    // ---- one-time prepack: W1 (f32, row-major [256,32]) -> bf16 B fragments ----
    for (int q = threadIdx.x; q < 8 * 2 * 2 * 32; q += 256) {
        int lane = q & 31;
        int lohi = (q >> 5) & 1;
        int c    = (q >> 6) & 1;
        int t    = q >> 7;
        int n    = c * 16 + (lane & 15);
        int kb   = t * 32 + (lane >> 4) * 8 + lohi * 16;   // + 2*jj below
        uint4 d;
        unsigned int* dw = &d.x;
        #pragma unroll
        for (int jj = 0; jj < 4; ++jj) {
            int k = kb + 2 * jj;
            dw[jj] = f2bf_bits(W1[k * 32 + n]) |
                     (f2bf_bits(W1[(k + 1) * 32 + n]) << 16);
        }
        sB[q] = d;
    }
    __syncthreads();

    const int lane = threadIdx.x & 31;
    const int wave = (blockIdx.x * 256 + threadIdx.x) >> 5;   // tile id (uniform per wave)
    if (wave * 16 >= NNODES) return;                          // wave-uniform exit

    const int m  = lane & 15;   // A row within tile / C column
    const int hf = lane >> 4;   // lane half selects K-subsets
    const float* __restrict__ xrow = x + (long)(wave * 16 + m) * FDIM;
    __builtin_prefetch(xrow + 128, 0, 1);                     // global_prefetch_b8

    v8f c0 = {}; v8f c1 = {};

    #pragma unroll 2
    for (int t = 0; t < 8; ++t) {
        const int k0 = t * 32;
        // ---- A fragment: 16-bit A 16x32 layout ----
        // lane(m,hf): elems 0..7  = A[m][k0 + hf*8 + 0..7]
        //             elems 8..15 = A[m][k0 + 16 + hf*8 + 0..7]
        float4 a0 = *reinterpret_cast<const float4*>(xrow + k0 + hf * 8);
        float4 a1 = *reinterpret_cast<const float4*>(xrow + k0 + hf * 8 + 4);
        float4 a2 = *reinterpret_cast<const float4*>(xrow + k0 + 16 + hf * 8);
        float4 a3 = *reinterpret_cast<const float4*>(xrow + k0 + 16 + hf * 8 + 4);
        v16bf a;
        a[0]  = (__bf16)a0.x; a[1]  = (__bf16)a0.y; a[2]  = (__bf16)a0.z; a[3]  = (__bf16)a0.w;
        a[4]  = (__bf16)a1.x; a[5]  = (__bf16)a1.y; a[6]  = (__bf16)a1.z; a[7]  = (__bf16)a1.w;
        a[8]  = (__bf16)a2.x; a[9]  = (__bf16)a2.y; a[10] = (__bf16)a2.z; a[11] = (__bf16)a2.w;
        a[12] = (__bf16)a3.x; a[13] = (__bf16)a3.y; a[14] = (__bf16)a3.z; a[15] = (__bf16)a3.w;

        // ---- B fragments from LDS (pre-packed, conflict-free b128 reads) ----
        BFrag32B f0, f1;
        f0.lo = sB[((t * 2 + 0) * 2 + 0) * 32 + lane];
        f0.hi = sB[((t * 2 + 0) * 2 + 1) * 32 + lane];
        f1.lo = sB[((t * 2 + 1) * 2 + 0) * 32 + lane];
        f1.hi = sB[((t * 2 + 1) * 2 + 1) * 32 + lane];
        v16bf b0 = __builtin_bit_cast(v16bf, f0);
        v16bf b1 = __builtin_bit_cast(v16bf, f1);

        c0 = __builtin_amdgcn_wmma_f32_16x16x32_bf16(false, a, false, b0,
                                                     (short)0, c0, false, false);
        c1 = __builtin_amdgcn_wmma_f32_16x16x32_bf16(false, a, false, b1,
                                                     (short)0, c1, false, false);
    }

    // ---- C store: 32-bit 16x16 C layout: VGPR r -> row hf*8 + r, col = lane&15 ----
    long base = (long)wave * 16;
    #pragma unroll
    for (int r = 0; r < 8; ++r) {
        h[(base + hf * 8 + r) * 32 + m]      = c0[r];
        h[(base + hf * 8 + r) * 32 + 16 + m] = c1[r];
    }
}

// ---------------- GCN scatter: agg[dst] += h[src] * dinv[src]*dinv[dst] ----------------
template <int D>
__global__ void __launch_bounds__(256)
scatter_kernel(const float* __restrict__ h, const int* __restrict__ ei,
               const float* __restrict__ dinv, float* __restrict__ agg) {
    long tid = (long)blockIdx.x * 256 + threadIdx.x;
    long e = tid / D;                 // D is power of 2 -> shifts
    int  f = (int)(tid & (D - 1));
    if (e >= NEDGES) return;
    int s = ei[e];
    int d = ei[NEDGES + e];
    float nrm = dinv[s] * dinv[d];    // broadcast within the D-lane group
    atomicAdd(&agg[(long)d * D + f], h[(long)s * D + f] * nrm);
}

// ---------------- self-loop + bias + tanh ----------------
template <int D>
__global__ void __launch_bounds__(256)
finalize_kernel(const float* __restrict__ h, const float* __restrict__ dinv,
                const float* __restrict__ b, float* __restrict__ agg) {
    long i = (long)blockIdx.x * 256 + threadIdx.x;
    if (i >= (long)NNODES * D) return;
    long n = i / D;
    int  f = (int)(i & (D - 1));
    float di = dinv[n];
    agg[i] = tanhf(agg[i] + h[i] * di * di + b[f]);
}

// ---------------- small dense layers: out = h @ W (+ bias) ----------------
template <int DIN, int DOUT, bool BIAS>
__global__ void __launch_bounds__(256)
gemm_small(const float* __restrict__ h, const float* __restrict__ W,
           const float* __restrict__ b, float* __restrict__ out) {
    __shared__ float sW[DIN * DOUT];
    __shared__ float sb[DOUT];
    for (int i = threadIdx.x; i < DIN * DOUT; i += 256) sW[i] = W[i];
    if (BIAS)
        for (int i = threadIdx.x; i < DOUT; i += 256) sb[i] = b[i];
    __syncthreads();

    int nid = blockIdx.x * 256 + threadIdx.x;
    if (nid >= NNODES) return;

    float row[DIN];
    #pragma unroll
    for (int k = 0; k < DIN; ++k) row[k] = h[(long)nid * DIN + k];
    #pragma unroll
    for (int j = 0; j < DOUT; ++j) {
        float acc = BIAS ? sb[j] : 0.0f;
        #pragma unroll
        for (int k = 0; k < DIN; ++k) acc += row[k] * sW[k * DOUT + j];
        out[(long)nid * DOUT + j] = acc;
    }
}

// ---------------- launch ----------------
static inline dim3 grid1(long n) { return dim3((unsigned)((n + 255) / 256)); }

extern "C" void kernel_launch(void* const* d_in, const int* in_sizes, int n_in,
                              void* d_out, int out_size, void* d_ws, size_t ws_size,
                              hipStream_t stream) {
    const float* x  = (const float*)d_in[0];
    const int*   ei = (const int*)  d_in[1];
    const float* W1 = (const float*)d_in[2];
    const float* b1 = (const float*)d_in[3];
    const float* W2 = (const float*)d_in[4];
    const float* b2 = (const float*)d_in[5];
    const float* W3 = (const float*)d_in[6];
    const float* b3 = (const float*)d_in[7];
    const float* Wc = (const float*)d_in[8];
    const float* bc = (const float*)d_in[9];
    float* out = (float*)d_out;

    // workspace layout: dinv[N] | bufA[N*32] | bufB[N*32]  (~52 MB)
    float* dinv = (float*)d_ws;
    float* bufA = dinv + NNODES;
    float* bufB = bufA + (size_t)NNODES * 32;

    dim3 blk(256);

    // degrees (with self-loop) -> dinv = rsqrt(deg+1)
    zero_kernel <<<grid1(NNODES), blk, 0, stream>>>(dinv, NNODES);
    deg_kernel  <<<grid1(NEDGES), blk, 0, stream>>>(ei, dinv);
    rsqrt_kernel<<<grid1(NNODES), blk, 0, stream>>>(dinv);

    // ---- layer 1: h1 = x@W1 (WMMA), conv, tanh ----
    gemm1_wmma<<<dim3((NNODES / 16 + 7) / 8), blk, 0, stream>>>(x, W1, bufA);
    zero_kernel       <<<grid1((long)NNODES * 32), blk, 0, stream>>>(bufB, (long)NNODES * 32);
    scatter_kernel<32><<<grid1((long)NEDGES * 32), blk, 0, stream>>>(bufA, ei, dinv, bufB);
    finalize_kernel<32><<<grid1((long)NNODES * 32), blk, 0, stream>>>(bufA, dinv, b1, bufB);

    // ---- layer 2 ----
    gemm_small<32, 16, false><<<grid1(NNODES), blk, 0, stream>>>(bufB, W2, nullptr, bufA);
    zero_kernel       <<<grid1((long)NNODES * 16), blk, 0, stream>>>(bufB, (long)NNODES * 16);
    scatter_kernel<16><<<grid1((long)NEDGES * 16), blk, 0, stream>>>(bufA, ei, dinv, bufB);
    finalize_kernel<16><<<grid1((long)NNODES * 16), blk, 0, stream>>>(bufA, dinv, b2, bufB);

    // ---- layer 3 ----
    gemm_small<16, 8, false><<<grid1(NNODES), blk, 0, stream>>>(bufB, W3, nullptr, bufA);
    zero_kernel      <<<grid1((long)NNODES * 8), blk, 0, stream>>>(bufB, (long)NNODES * 8);
    scatter_kernel<8><<<grid1((long)NEDGES * 8), blk, 0, stream>>>(bufA, ei, dinv, bufB);
    finalize_kernel<8><<<grid1((long)NNODES * 8), blk, 0, stream>>>(bufA, dinv, b3, bufB);

    // ---- classifier: out = h3 @ Wc + bc ----
    gemm_small<8, 4, true><<<grid1(NNODES), blk, 0, stream>>>(bufB, Wc, bc, out);
}
